// Sequence_82652350644954
// MI455X (gfx1250) — matrix operand order repaired and checked by reading, not compiled, not run on hardware
//
#include <hip/hip_runtime.h>
#include <hip/hip_bf16.h>

// Problem constants (match reference)
#define B_   64
#define T_   256
#define E_   512
#define H_   1024
#define V_   5000
#define VPAD 5120      // V padded to multiple of 64 for clean tiling
#define G4H  4096      // 4*H

typedef __attribute__((ext_vector_type(16))) _Float16 v16h;
typedef __attribute__((ext_vector_type(8)))  _Float16 v8h;
typedef __attribute__((ext_vector_type(8)))  float    v8f;
typedef __attribute__((ext_vector_type(4)))  int      v4i;

// Address-space-qualified pointee types for the gfx1250 async-LDS builtins
// (signature per compiler diagnostic: (v4i __device__*, v4i __shared__*, Ii, Ii)).
typedef __attribute__((address_space(1))) v4i as1_v4i;
typedef __attribute__((address_space(3))) v4i as3_v4i;

// ---------------------------------------------------------------------------
// gfx1250 async global->LDS copy (ASYNCcnt-tracked). Guarded so the file
// still compiles if the toolchain lacks the builtins (sync ds_store fallback).
#if defined(__has_builtin)
#if __has_builtin(__builtin_amdgcn_global_load_async_to_lds_b128) && \
    __has_builtin(__builtin_amdgcn_s_wait_asynccnt)
#define USE_ASYNC_LDS 1
#endif
#endif

__device__ __forceinline__ void cp16_lds(void* lds_dst, const void* gsrc) {
#ifdef USE_ASYNC_LDS
  __builtin_amdgcn_global_load_async_to_lds_b128(
      (as1_v4i*)gsrc, (as3_v4i*)lds_dst, 0, 0);
#else
  *(uint4*)lds_dst = *(const uint4*)gsrc;   // sync fallback via ds_store_b128
#endif
}

template <int N>
__device__ __forceinline__ void wait_async() {
#ifdef USE_ASYNC_LDS
  __builtin_amdgcn_s_wait_asynccnt(N);
#endif
}

// ---------------------------------------------------------------------------
// CDNA5 wave32 WMMA fragment loaders (16x16x32 f16 -> f32)
//
// A (16Mx32K), 16-bit: lane l (l<16) holds M=l, K={0..7,16..23}; lane l+16
// holds M=l, K={8..15,24..31}. Two contiguous 16B chunks per lane.
__device__ __forceinline__ v16h load_frag_a(const _Float16* __restrict__ p,
                                            int ld, int k0) {
  const int l    = threadIdx.x & 31;
  const int m    = l & 15;
  const int half = l >> 4;
  const _Float16* r = p + (size_t)m * ld + k0 + half * 8;
  v8h lo = *(const v8h*)(r);
  v8h hi = *(const v8h*)(r + 16);
  v16h out;
#pragma unroll
  for (int i = 0; i < 8; ++i) { out[i] = lo[i]; out[i + 8] = hi[i]; }
  return out;
}

// B (32Kx16N) fragment out of an LDS tile laid out as rows of 32 halves:
// lane l holds column n=l&15, K={(l>>4)*8..+7, 16+(l>>4)*8..+7} (two 16B chunks).
__device__ __forceinline__ v16h load_frag_b_smem(const _Float16* base,
                                                 int row0) {
  const int l  = threadIdx.x & 31;
  const int n  = l & 15;
  const int kh = l >> 4;
  const _Float16* r = base + (row0 + n) * 32 + kh * 8;
  v8h lo = *(const v8h*)(r);
  v8h hi = *(const v8h*)(r + 16);
  v16h out;
#pragma unroll
  for (int i = 0; i < 8; ++i) { out[i] = lo[i]; out[i + 8] = hi[i]; }
  return out;
}

__device__ __forceinline__ float sigmoidf_(float x) {
  return 1.0f / (1.0f + __expf(-x));
}

// ---------------------------------------------------------------------------
// One LSTM layer, one timestep.  gates = bias_a + bias_b + A1@W1^T + A2@W2^T
// A1:(B,K1) f16, W1:(4H,K1) f16, A2:(B,H) f16, W2:(4H,H) f16.
// Each wave computes one 16(M)x16(col) tile for all 4 gates simultaneously,
// so the cell nonlinearity is pure in-register.  The 128 W rows a block needs
// per 32-K chunk (4 gates x 32 cols) are async-staged into LDS once and
// shared by all 8 waves (4x traffic cut), double-buffered against WMMA.
// Grid: H/32 blocks, 256 threads (8 waves = 4 M-tiles x 2 col-tiles).
__global__ void __launch_bounds__(256)
lstm_step_kernel(const _Float16* __restrict__ A1,
                 const _Float16* __restrict__ W1, int K1,
                 const _Float16* __restrict__ A2,
                 const _Float16* __restrict__ W2,
                 const float* __restrict__ bias_a,
                 const float* __restrict__ bias_b,
                 float* __restrict__ c_state,
                 _Float16* __restrict__ h_out) {
  __shared__ __align__(64) _Float16 sW[2][128 * 32];   // 2 x 8 KB

  const int wave  = threadIdx.x >> 5;
  const int lane  = threadIdx.x & 31;
  const int mtile = (wave & 3) * 16;          // batch rows 0..63
  const int ct    = (wave >> 2) * 16;         // col sub-tile within block
  const int c0b   = blockIdx.x * 32;          // block's hidden col base
  const int ncol  = c0b + ct + (lane & 15);

  v8f acc[4];
#pragma unroll
  for (int g = 0; g < 4; ++g) {
    const float bv = bias_a[g * H_ + ncol] + bias_b[g * H_ + ncol];
#pragma unroll
    for (int j = 0; j < 8; ++j) acc[g][j] = bv;
  }

  // Stage one 32-K chunk of 128 W rows (gate-major) into sW[buf].
  // 512 x 16B segments -> 2 per thread -> 2 async instructions per wave.
  auto stageW = [&](const _Float16* Wb, int ld, int buf, int k0) {
#pragma unroll
    for (int s = 0; s < 2; ++s) {
      const int rr  = (threadIdx.x >> 2) + s * 64;   // 0..127
      const int seg = (threadIdx.x & 3) * 8;         // halves (16B units)
      const int g = rr >> 5, cc = rr & 31;
      cp16_lds(&sW[buf][rr * 32 + seg],
               Wb + (size_t)(g * H_ + c0b + cc) * ld + k0 + seg);
    }
  };

  auto gemm_phase = [&](const _Float16* A, const _Float16* W, int ld) {
    const int NK = ld / 32;
    const _Float16* ap = A + (size_t)mtile * ld;
    stageW(W, ld, 0, 0);
    stageW(W, ld, 1, 32);
    for (int kc = 0; kc < NK; ++kc) {
      if (kc + 1 < NK) wait_async<2>(); else wait_async<0>();
      __syncthreads();
      v16h af = load_frag_a(ap, ld, kc * 32);
      const _Float16* sb = &sW[kc & 1][0];
#pragma unroll
      for (int g = 0; g < 4; ++g) {
        v16h bf = load_frag_b_smem(sb, g * 32 + ct);
        acc[g] = __builtin_amdgcn_wmma_f32_16x16x32_f16(
            false, af, false, bf, (short)0, acc[g], false, false);
      }
      __syncthreads();
      if (kc + 2 < NK) stageW(W, ld, kc & 1, (kc + 2) * 32);
    }
  };

  gemm_phase(A1, W1, K1);   // input contribution
  gemm_phase(A2, W2, H_);   // recurrent contribution

  // LSTM cell elementwise (f32); gate tiles aligned by construction.
#pragma unroll
  for (int j = 0; j < 8; ++j) {
    const int m = mtile + (lane >> 4) * 8 + j;       // C/D layout row
    const size_t idx = (size_t)m * H_ + ncol;
    const float iv = sigmoidf_(acc[0][j]);
    const float fv = sigmoidf_(acc[1][j]);
    const float gv = tanhf(acc[2][j]);
    const float ov = sigmoidf_(acc[3][j]);
    const float cn = fv * c_state[idx] + iv * gv;
    c_state[idx] = cn;
    h_out[idx]   = (_Float16)(ov * tanhf(cn));
  }
}

// ---------------------------------------------------------------------------
// Final projection: out[b,t,v] = lin_b[v] + H2[(t*B+b),:] . lin_W[v,:]
// M = T*B = 16384, N = VPAD = 5120, K = H.  Wave: 16Mx64N; block: 8 waves
// covering 128Mx64N; the 64 lin_W rows per 32-K chunk are async-staged into
// LDS once per block (8x traffic cut).  Grid: (128, 80).
__global__ void __launch_bounds__(256)
proj_kernel(const _Float16* __restrict__ H2,
            const _Float16* __restrict__ Wl,
            const float* __restrict__ lin_b,
            float* __restrict__ out) {
  __shared__ __align__(64) _Float16 sB[2][64 * 32];   // 2 x 4 KB

  const int wave = threadIdx.x >> 5;
  const int lane = threadIdx.x & 31;
  const int r0   = blockIdx.x * 128 + wave * 16;
  const int v0   = blockIdx.y * 64;

  const int srow = threadIdx.x >> 2;        // 0..63
  const int sseg = (threadIdx.x & 3) * 8;   // halves (16B units)
  const _Float16* gB = Wl + (size_t)(v0 + srow) * H_ + sseg;

  auto stage = [&](int buf, int k0) {
    cp16_lds(&sB[buf][srow * 32 + sseg], gB + k0);   // 1 async op per wave
  };

  v8f acc[4];
#pragma unroll
  for (int q = 0; q < 4; ++q)
#pragma unroll
    for (int j = 0; j < 8; ++j) acc[q][j] = 0.0f;

  const _Float16* ap = H2 + (size_t)r0 * H_;
  const int NK = H_ / 32;   // 32
  stage(0, 0);
  stage(1, 32);
  for (int kc = 0; kc < NK; ++kc) {
    if (kc + 1 < NK) wait_async<1>(); else wait_async<0>();
    __syncthreads();
    v16h af = load_frag_a(ap, H_, kc * 32);
    const _Float16* sb = &sB[kc & 1][0];
#pragma unroll
    for (int q = 0; q < 4; ++q) {
      v16h bf = load_frag_b_smem(sb, q * 16);
      acc[q] = __builtin_amdgcn_wmma_f32_16x16x32_f16(
          false, af, false, bf, (short)0, acc[q], false, false);
    }
    __syncthreads();
    if (kc + 2 < NK) stage(kc & 1, (kc + 2) * 32);
  }

#pragma unroll
  for (int q = 0; q < 4; ++q) {
    const int v = v0 + q * 16 + (lane & 15);
    if (v < V_) {
      const float bb = lin_b[v];
#pragma unroll
      for (int j = 0; j < 8; ++j) {
        const int r  = r0 + (lane >> 4) * 8 + j;  // r = t*B + b
        const int tt = r >> 6;                    // / B_
        const int b  = r & 63;
        out[((size_t)b * T_ + tt) * V_ + v] = acc[q][j] + bb;
      }
    }
  }
}

// ---------------------------------------------------------------------------
// Utility kernels
__global__ void cvt_f32_f16(const float* __restrict__ src,
                            _Float16* __restrict__ dst, int n_src, int n_dst) {
  const int i = blockIdx.x * blockDim.x + threadIdx.x;
  if (i < n_dst) dst[i] = (i < n_src) ? (_Float16)src[i] : (_Float16)0.0f;
}

// embed gather + f16 convert; xe layout (T,B,E)
__global__ void gather_embed(const int* __restrict__ tok,
                             const float* __restrict__ embW,
                             _Float16* __restrict__ xe) {
  const int bid = blockIdx.x;          // b*T + t  (input is (B,T) row-major)
  const int b = bid >> 8;              // T_ == 256
  const int t = bid & 255;
  const int token = tok[bid];
  const float* src = embW + (size_t)token * E_;
  _Float16* dst = xe + ((size_t)t * B_ + b) * E_;
  for (int e = threadIdx.x; e < E_; e += blockDim.x)
    dst[e] = (_Float16)src[e];
}

__global__ void zero_ws(uint4* __restrict__ p, int n16) {
  const int i = blockIdx.x * blockDim.x + threadIdx.x;
  if (i < n16) p[i] = make_uint4(0u, 0u, 0u, 0u);
}

// ---------------------------------------------------------------------------
extern "C" void kernel_launch(void* const* d_in, const int* in_sizes, int n_in,
                              void* d_out, int out_size, void* d_ws,
                              size_t ws_size, hipStream_t stream) {
  (void)in_sizes; (void)n_in; (void)out_size; (void)ws_size;
  const int*   input = (const int*)  d_in[0];
  // d_in[1] = future (scalar 0, ignored)
  const float* embW  = (const float*)d_in[2];
  const float* W_ih1 = (const float*)d_in[3];
  const float* W_hh1 = (const float*)d_in[4];
  const float* b_ih1 = (const float*)d_in[5];
  const float* b_hh1 = (const float*)d_in[6];
  const float* W_ih2 = (const float*)d_in[7];
  const float* W_hh2 = (const float*)d_in[8];
  const float* b_ih2 = (const float*)d_in[9];
  const float* b_hh2 = (const float*)d_in[10];
  const float* lin_W = (const float*)d_in[11];
  const float* lin_b = (const float*)d_in[12];
  float* out = (float*)d_out;

  // ---- workspace carving (~88 MB) ----
  char* w = (char*)d_ws;
  size_t off = 0;
  auto carve = [&](size_t bytes) -> void* {
    void* p = w + off;
    off = (off + bytes + 255) & ~(size_t)255;
    return p;
  };
  _Float16* Wih1h = (_Float16*)carve((size_t)G4H * E_ * 2);
  _Float16* Whh1h = (_Float16*)carve((size_t)G4H * H_ * 2);
  _Float16* Wih2h = (_Float16*)carve((size_t)G4H * H_ * 2);
  _Float16* Whh2h = (_Float16*)carve((size_t)G4H * H_ * 2);
  _Float16* linWh = (_Float16*)carve((size_t)VPAD * H_ * 2);
  _Float16* xe    = (_Float16*)carve((size_t)T_ * B_ * E_ * 2);
  _Float16* H2    = (_Float16*)carve((size_t)T_ * B_ * H_ * 2);
  const size_t stateBytes = (size_t)3 * B_ * H_ * 2 + (size_t)2 * B_ * H_ * 4;
  char* stateBase = (char*)carve(stateBytes);
  _Float16* h1a = (_Float16*)stateBase;
  _Float16* h1b = h1a + (size_t)B_ * H_;
  _Float16* h2z = h1b + (size_t)B_ * H_;   // zero h2 for t==0
  float* c1 = (float*)(h2z + (size_t)B_ * H_);
  float* c2 = c1 + (size_t)B_ * H_;

  const int TPB = 256;
  auto nblk = [](size_t n) { return (unsigned)((n + 255) / 256); };

  // ---- one-time-per-call prep (deterministic; no state across calls) ----
  cvt_f32_f16<<<nblk((size_t)G4H * E_), TPB, 0, stream>>>(W_ih1, Wih1h, G4H * E_, G4H * E_);
  cvt_f32_f16<<<nblk((size_t)G4H * H_), TPB, 0, stream>>>(W_hh1, Whh1h, G4H * H_, G4H * H_);
  cvt_f32_f16<<<nblk((size_t)G4H * H_), TPB, 0, stream>>>(W_ih2, Wih2h, G4H * H_, G4H * H_);
  cvt_f32_f16<<<nblk((size_t)G4H * H_), TPB, 0, stream>>>(W_hh2, Whh2h, G4H * H_, G4H * H_);
  cvt_f32_f16<<<nblk((size_t)VPAD * H_), TPB, 0, stream>>>(lin_W, linWh, V_ * H_, VPAD * H_);
  gather_embed<<<B_ * T_, TPB, 0, stream>>>(input, embW, xe);
  zero_ws<<<nblk(stateBytes / 16), TPB, 0, stream>>>((uint4*)stateBase,
                                                     (int)(stateBytes / 16));

  // ---- sequential recurrence: one launch per (step, layer) ----
  for (int t = 0; t < T_; ++t) {
    const _Float16* h1_in  = (t & 1) ? h1b : h1a;   // ping-pong (race-free)
    _Float16*       h1_out = (t & 1) ? h1a : h1b;
    lstm_step_kernel<<<H_ / 32, TPB, 0, stream>>>(
        xe + (size_t)t * B_ * E_, Wih1h, E_, h1_in, Whh1h,
        b_ih1, b_hh1, c1, h1_out);

    const _Float16* h2_in = t ? (H2 + (size_t)(t - 1) * B_ * H_) : h2z;
    lstm_step_kernel<<<H_ / 32, TPB, 0, stream>>>(
        h1_out, Wih2h, H_, h2_in, Whh2h,
        b_ih2, b_hh2, c2, H2 + (size_t)t * B_ * H_);
  }

  // ---- batched output projection (dominant WMMA GEMM, 172 GFLOP) ----
  proj_kernel<<<dim3((T_ * B_) / 128, VPAD / 64), TPB, 0, stream>>>(
      H2, linWh, lin_b, out);
}
